// CreditRiskEnsemble_64192581206383
// MI455X (gfx1250) — compile-verified, matrix operand order checked by reading, not run on hardware
//
#include <hip/hip_runtime.h>
#include <hip/hip_bf16.h>

#define N_NODES 50000
#define N_EDGES 400000
#define ETOT (N_NODES + N_EDGES)
#define F_IN 66
#define HID 128
#define HEADS 8
#define NEG_SLOPE 0.2f

typedef __attribute__((ext_vector_type(16))) _Float16 v16h;
typedef __attribute__((ext_vector_type(8)))  _Float16 v8h;
typedef __attribute__((ext_vector_type(8)))  float    v8f;

// ---------- helpers ----------
__device__ __forceinline__ unsigned fenc(float f) {
  unsigned u = __float_as_uint(f);
  return (u & 0x80000000u) ? ~u : (u | 0x80000000u);   // monotone float->uint
}
__device__ __forceinline__ float fdec(unsigned u) {
  return __uint_as_float((u & 0x80000000u) ? (u & 0x7FFFFFFFu) : ~u);
}
__device__ __forceinline__ void edge_rc(const int* __restrict__ ei, int eid, int& r, int& c) {
  if (eid < N_EDGES) { r = ei[eid]; c = ei[N_EDGES + eid]; }
  else               { r = c = eid - N_EDGES; }          // self loop
}

// ---------- fills ----------
__global__ void k_fill_f32(float* p, float v, size_t n) {
  size_t i = (size_t)blockIdx.x * blockDim.x + threadIdx.x;
  size_t st = (size_t)gridDim.x * blockDim.x;
  for (; i < n; i += st) p[i] = v;
}
__global__ void k_fill_u32(unsigned* p, unsigned v, size_t n) {
  size_t i = (size_t)blockIdx.x * blockDim.x + threadIdx.x;
  size_t st = (size_t)gridDim.x * blockDim.x;
  for (; i < n; i += st) p[i] = v;
}

// ---------- degree / normalization ----------
__global__ void k_deg(const int* __restrict__ ei, float* deg) {
  int e = blockIdx.x * blockDim.x + threadIdx.x;
  if (e < N_EDGES) atomicAdd(&deg[ei[N_EDGES + e]], 1.0f);
}
__global__ void k_dinv(float* deg) {
  int i = blockIdx.x * blockDim.x + threadIdx.x;
  if (i < N_NODES) deg[i] = rsqrtf(fmaxf(deg[i], 1.0f));
}

// ---------- precision prep ----------
// activations f32[nrows,K] -> f16[nrows,Kp] with zero pad (Kp multiple of 32)
__global__ void k_cvt_f16_pad(const float* __restrict__ in, _Float16* __restrict__ out,
                              int nrows, int K, int Kp) {
  size_t total = (size_t)nrows * Kp;
  size_t i = (size_t)blockIdx.x * blockDim.x + threadIdx.x;
  size_t st = (size_t)gridDim.x * blockDim.x;
  for (; i < total; i += st) {
    int r = (int)(i / Kp), k = (int)(i % Kp);
    out[i] = (k < K) ? (_Float16)in[(size_t)r * K + k] : (_Float16)0.f;
  }
}
// weights f32[K,C] -> f16 Wt[C,Kp] (transposed, zero pad)
__global__ void k_w_transpose(const float* __restrict__ W, _Float16* __restrict__ Wt,
                              int K, int C, int Kp) {
  size_t total = (size_t)C * Kp;
  size_t i = (size_t)blockIdx.x * blockDim.x + threadIdx.x;
  size_t st = (size_t)gridDim.x * blockDim.x;
  for (; i < total; i += st) {
    int c = (int)(i / Kp), k = (int)(i % Kp);
    Wt[i] = (k < K) ? (_Float16)W[(size_t)k * C + c] : (_Float16)0.f;
  }
}

// ---------- WMMA GEMM: C[nrows,ncols](f16) = A[nrows,Kp](f16) @ Wt[ncols,Kp]^T ----------
// block = 256 threads = 8 waves; tile 64x64; wave computes 16x32 (2 wmma tiles).
// Per-lane fragment K layout: k = 16*(j/8) + 8*half + (j%8)  ->  two contiguous
// 8-half (16B) chunks per fragment -> 2x global_load_b128, no LDS, no guards.
__global__ __launch_bounds__(256) void k_gemm_wmma(
    const _Float16* __restrict__ A, const _Float16* __restrict__ Wt,
    _Float16* __restrict__ C, int nrows, int Kp, int ncols) {
  const int lane = threadIdx.x & 31;
  const int wv   = threadIdx.x >> 5;
  const int wr   = wv >> 1;        // 0..3 : 16-row sub-tile
  const int wc   = wv & 1;         // 0..1 : 32-col sub-tile
  const int row0 = blockIdx.y * 64;
  const int col0 = blockIdx.x * 64;
  const int m    = lane & 15;
  const int hf   = lane >> 4;

  int arow = row0 + wr * 16 + m;
  if (arow >= nrows) arow = nrows - 1;     // clamp: polluted rows are never stored
  const _Float16* ap  = A  + (size_t)arow * Kp + hf * 8;
  const _Float16* bp0 = Wt + (size_t)(col0 + wc * 32 + m) * Kp + hf * 8;
  const _Float16* bp1 = bp0 + (size_t)16 * Kp;

  v8f acc0 = {}; v8f acc1 = {};
  for (int k0 = 0; k0 < Kp; k0 += 32) {
    v8h a_lo = *(const v8h*)(ap  + k0);
    v8h a_hi = *(const v8h*)(ap  + k0 + 16);
    v8h b0lo = *(const v8h*)(bp0 + k0);
    v8h b0hi = *(const v8h*)(bp0 + k0 + 16);
    v8h b1lo = *(const v8h*)(bp1 + k0);
    v8h b1hi = *(const v8h*)(bp1 + k0 + 16);
    v16h af  = __builtin_shufflevector(a_lo, a_hi, 0,1,2,3,4,5,6,7,8,9,10,11,12,13,14,15);
    v16h bf0 = __builtin_shufflevector(b0lo, b0hi, 0,1,2,3,4,5,6,7,8,9,10,11,12,13,14,15);
    v16h bf1 = __builtin_shufflevector(b1lo, b1hi, 0,1,2,3,4,5,6,7,8,9,10,11,12,13,14,15);
    acc0 = __builtin_amdgcn_wmma_f32_16x16x32_f16(false, af, false, bf0, (short)0, acc0, false, false);
    acc1 = __builtin_amdgcn_wmma_f32_16x16x32_f16(false, af, false, bf1, (short)0, acc1, false, false);
  }
  // C/D layout: VGPR r -> M = r + 8*half; lane%16 -> N
  #pragma unroll
  for (int r = 0; r < 8; ++r) {
    int gr = row0 + wr * 16 + hf * 8 + r;
    if (gr < nrows) {
      int gc = col0 + wc * 32 + m;
      C[(size_t)gr * ncols + gc]      = (_Float16)acc0[r];
      C[(size_t)gr * ncols + gc + 16] = (_Float16)acc1[r];
    }
  }
}

// ---------- GCN aggregation: agg[col] += dinv[row]*dinv[col] * h[row] ----------
// wave per edge; each lane owns a contiguous C/32 channel chunk (coalesced).
__global__ void k_gcn_agg(const int* __restrict__ ei, const _Float16* __restrict__ h,
                          const float* __restrict__ dinv, float* __restrict__ agg, int C) {
  int gw   = blockIdx.x * (blockDim.x >> 5) + (threadIdx.x >> 5);
  int lane = threadIdx.x & 31;
  if (gw >= ETOT) return;
  int r, c; edge_rc(ei, gw, r, c);
  float w = dinv[r] * dinv[c];
  int chunk = C >> 5;
  const _Float16* hp = h + (size_t)r * C + lane * chunk;
  float* op = agg + (size_t)c * C + lane * chunk;
  for (int t = 0; t < chunk; ++t)
    atomicAdd(&op[t], w * (float)hp[t]);
}

// ---------- bias + activation (in place): 1=relu, 2=elu ----------
__global__ void k_bias_act(float* buf, const float* __restrict__ b, size_t total, int C, int act) {
  size_t i = (size_t)blockIdx.x * blockDim.x + threadIdx.x;
  size_t st = (size_t)gridDim.x * blockDim.x;
  for (; i < total; i += st) {
    float v = buf[i] + b[i % (size_t)C];
    if (act == 1)      v = fmaxf(v, 0.f);
    else if (act == 2) v = (v > 0.f) ? v : expm1f(v);
    buf[i] = v;
  }
}

// ---------- GAT: per-node attention scores ----------
__global__ void k_gat_scores(const _Float16* __restrict__ h, const float* __restrict__ a_src,
                             const float* __restrict__ a_dst, float* asrc, float* adst,
                             int Hd, int C) {
  int idx = blockIdx.x * blockDim.x + threadIdx.x;     // idx = n*Hd + hd
  if (idx >= N_NODES * Hd) return;
  int hd = idx % Hd;
  const _Float16* hp = h + (size_t)idx * C;
  const float* as = a_src + hd * C;
  const float* ad = a_dst + hd * C;
  float s1 = 0.f, s2 = 0.f;
  for (int ch = 0; ch < C; ++ch) {
    float v = (float)hp[ch];
    s1 += v * as[ch]; s2 += v * ad[ch];
  }
  asrc[idx] = s1; adst[idx] = s2;
}

// ---------- GAT: segment max over incoming edges (encoded-uint atomicMax) ----------
__global__ void k_gat_emax(const int* __restrict__ ei, const float* __restrict__ asrc,
                           const float* __restrict__ adst, unsigned* mmax, int Hd) {
  long long idx = (long long)blockIdx.x * blockDim.x + threadIdx.x;
  if (idx >= (long long)ETOT * Hd) return;
  int eid = (int)(idx / Hd), hd = (int)(idx % Hd);
  int r, c; edge_rc(ei, eid, r, c);
  float e = asrc[r * Hd + hd] + adst[c * Hd + hd];
  e = (e > 0.f) ? e : NEG_SLOPE * e;
  atomicMax(&mmax[c * Hd + hd], fenc(e));
}

// ---------- GAT: exp + segment sum ----------
__global__ void k_gat_esum(const int* __restrict__ ei, const float* __restrict__ asrc,
                           const float* __restrict__ adst, const unsigned* __restrict__ mmax,
                           float* ex, float* denom, int Hd) {
  long long idx = (long long)blockIdx.x * blockDim.x + threadIdx.x;
  if (idx >= (long long)ETOT * Hd) return;
  int eid = (int)(idx / Hd), hd = (int)(idx % Hd);
  int r, c; edge_rc(ei, eid, r, c);
  float e = asrc[r * Hd + hd] + adst[c * Hd + hd];
  e = (e > 0.f) ? e : NEG_SLOPE * e;
  float ev = expf(e - fdec(mmax[c * Hd + hd]));
  ex[idx] = ev;
  atomicAdd(&denom[c * Hd + hd], ev);
}

// ---------- GAT: weighted message aggregation (wave per edge-head) ----------
__global__ void k_gat_msg(const int* __restrict__ ei, const _Float16* __restrict__ h,
                          const float* __restrict__ ex, const float* __restrict__ denom,
                          float* __restrict__ agg, int Hd, int C) {
  long long gw = (long long)blockIdx.x * (blockDim.x >> 5) + (threadIdx.x >> 5);
  int lane = threadIdx.x & 31;
  if (gw >= (long long)ETOT * Hd) return;
  int eid = (int)(gw / Hd), hd = (int)(gw % Hd);
  int r, c; edge_rc(ei, eid, r, c);
  float alpha = ex[gw] / denom[c * Hd + hd];
  int chunk = C >> 5;
  const _Float16* hp = h + ((size_t)r * Hd + hd) * C + lane * chunk;
  float* op = agg + ((size_t)c * Hd + hd) * C + lane * chunk;
  for (int t = 0; t < chunk; ++t)
    atomicAdd(&op[t], alpha * (float)hp[t]);
}

// ---------- readout: sigmoid(feat @ Wr + br) ----------
__global__ void k_readout(const float* __restrict__ feat, const float* __restrict__ Wr,
                          const float* __restrict__ br, float* out, int C) {
  int n = blockIdx.x * blockDim.x + threadIdx.x;
  if (n >= N_NODES) return;
  float s = br[0];
  const float* f = feat + (size_t)n * C;
  for (int ch = 0; ch < C; ++ch) s += f[ch] * Wr[ch];
  out[n] = 1.f / (1.f + expf(-s));
}

// ---------- ensemble ----------
__global__ void k_ensemble(const float* __restrict__ g, const float* __restrict__ a,
                           const float* __restrict__ W, const float* __restrict__ b,
                           float* out) {
  int n = blockIdx.x * blockDim.x + threadIdx.x;
  if (n >= N_NODES) return;
  float s = g[n] * W[0] + a[n] * W[1] + b[0];
  out[n] = 1.f / (1.f + expf(-s));
}

// =====================================================================
extern "C" void kernel_launch(void* const* d_in, const int* in_sizes, int n_in,
                              void* d_out, int out_size, void* d_ws, size_t ws_size,
                              hipStream_t stream) {
  (void)in_sizes; (void)n_in; (void)out_size; (void)ws_size;
  const float* x      = (const float*)d_in[0];
  const int*   ei     = (const int*)d_in[1];
  const float* gcn_W1 = (const float*)d_in[2];  const float* gcn_b1 = (const float*)d_in[3];
  const float* gcn_W2 = (const float*)d_in[4];  const float* gcn_b2 = (const float*)d_in[5];
  const float* gcn_W3 = (const float*)d_in[6];  const float* gcn_b3 = (const float*)d_in[7];
  const float* gcn_Wr = (const float*)d_in[8];  const float* gcn_br = (const float*)d_in[9];
  const float* gat_W1 = (const float*)d_in[10];
  const float* gat_as1 = (const float*)d_in[11]; const float* gat_ad1 = (const float*)d_in[12];
  const float* gat_b1 = (const float*)d_in[13];
  const float* gat_W2 = (const float*)d_in[14];
  const float* gat_as2 = (const float*)d_in[15]; const float* gat_ad2 = (const float*)d_in[16];
  const float* gat_b2 = (const float*)d_in[17];
  const float* gat_W3 = (const float*)d_in[18];
  const float* gat_as3 = (const float*)d_in[19]; const float* gat_ad3 = (const float*)d_in[20];
  const float* gat_b3 = (const float*)d_in[21];
  const float* gat_Wr = (const float*)d_in[22]; const float* gat_br = (const float*)d_in[23];
  const float* ens_W  = (const float*)d_in[24]; const float* ens_b  = (const float*)d_in[25];
  float* out = (float*)d_out;

  // ---- workspace carve ----
  char* p = (char*)d_ws;
  auto carve = [&](size_t bytes) -> void* {
    void* r = (void*)p;
    p += (bytes + 255) & ~(size_t)255;
    return r;
  };
  float*    dinv   = (float*)carve(sizeof(float) * N_NODES);
  _Float16* Ain    = (_Float16*)carve(sizeof(_Float16) * (size_t)N_NODES * 1024);
  _Float16* hbuf   = (_Float16*)carve(sizeof(_Float16) * (size_t)N_NODES * 1024);
  _Float16* Wt     = (_Float16*)carve(sizeof(_Float16) * 131072);   // max 128x1024
  float*    agg    = (float*)carve(sizeof(float) * (size_t)N_NODES * 1024);
  float*    asrc   = (float*)carve(sizeof(float) * N_NODES * HEADS);
  float*    adst   = (float*)carve(sizeof(float) * N_NODES * HEADS);
  unsigned* mmax   = (unsigned*)carve(sizeof(unsigned) * N_NODES * HEADS);
  float*    denom  = (float*)carve(sizeof(float) * N_NODES * HEADS);
  float*    ex     = (float*)carve(sizeof(float) * (size_t)ETOT * HEADS);
  float*    gcnout = (float*)carve(sizeof(float) * N_NODES);
  float*    gatout = (float*)carve(sizeof(float) * N_NODES);

  const int TB = 256;

  // ---- degrees -> dinv ----
  k_fill_f32<<<512, TB, 0, stream>>>(dinv, 1.0f, (size_t)N_NODES);   // self loop counts
  k_deg<<<(N_EDGES + TB - 1) / TB, TB, 0, stream>>>(ei, dinv);
  k_dinv<<<(N_NODES + TB - 1) / TB, TB, 0, stream>>>(dinv);

  // ---- GEMM = convert A (f16, padded K) + transpose W (f16) + WMMA ----
  auto gemm = [&](const float* Af32, const float* W, int K, int ncols) {
    int Kp = (K + 31) & ~31;
    k_cvt_f16_pad<<<2048, TB, 0, stream>>>(Af32, Ain, N_NODES, K, Kp);
    k_w_transpose<<<256, TB, 0, stream>>>(W, Wt, K, ncols, Kp);
    dim3 grid(ncols / 64, (N_NODES + 63) / 64);
    k_gemm_wmma<<<grid, 256, 0, stream>>>(Ain, Wt, hbuf, N_NODES, Kp, ncols);
  };

  // ---- GCN branch ----
  auto gcn_layer = [&](const float* A, const float* W, const float* b, int K, int C) {
    gemm(A, W, K, C);
    k_fill_f32<<<2048, TB, 0, stream>>>(agg, 0.f, (size_t)N_NODES * C);
    k_gcn_agg<<<(ETOT + 7) / 8, TB, 0, stream>>>(ei, hbuf, dinv, agg, C);
    k_bias_act<<<2048, TB, 0, stream>>>(agg, b, (size_t)N_NODES * C, C, 1);
  };
  gcn_layer(x,   gcn_W1, gcn_b1, F_IN, HID);
  gcn_layer(agg, gcn_W2, gcn_b2, HID,  HID);
  gcn_layer(agg, gcn_W3, gcn_b3, HID,  HID / 2);
  k_readout<<<(N_NODES + TB - 1) / TB, TB, 0, stream>>>(agg, gcn_Wr, gcn_br, gcnout, HID / 2);

  // ---- GAT branch ----
  auto gat_layer = [&](const float* A, const float* W, const float* as, const float* ad,
                       const float* b, int K, int Hd, int C) {
    gemm(A, W, K, Hd * C);
    int NH = N_NODES * Hd;
    k_gat_scores<<<(NH + TB - 1) / TB, TB, 0, stream>>>(hbuf, as, ad, asrc, adst, Hd, C);
    k_fill_u32<<<512, TB, 0, stream>>>(mmax, 0x007FFFFFu, (size_t)NH);  // fenc(-inf)
    long long EH = (long long)ETOT * Hd;
    int eb = (int)((EH + TB - 1) / TB);
    k_gat_emax<<<eb, TB, 0, stream>>>(ei, asrc, adst, mmax, Hd);
    k_fill_f32<<<512, TB, 0, stream>>>(denom, 0.f, (size_t)NH);
    k_gat_esum<<<eb, TB, 0, stream>>>(ei, asrc, adst, mmax, ex, denom, Hd);
    k_fill_f32<<<2048, TB, 0, stream>>>(agg, 0.f, (size_t)N_NODES * Hd * C);
    int mb = (int)((EH + 7) / 8);
    k_gat_msg<<<mb, TB, 0, stream>>>(ei, hbuf, ex, denom, agg, Hd, C);
    k_bias_act<<<2048, TB, 0, stream>>>(agg, b, (size_t)N_NODES * Hd * C, Hd * C, 2);
  };
  gat_layer(x,   gat_W1, gat_as1, gat_ad1, gat_b1, F_IN,        HEADS, HID);
  gat_layer(agg, gat_W2, gat_as2, gat_ad2, gat_b2, HEADS * HID, 1,     HID);
  gat_layer(agg, gat_W3, gat_as3, gat_ad3, gat_b3, HID,         1,     HID / 2);
  k_readout<<<(N_NODES + TB - 1) / TB, TB, 0, stream>>>(agg, gat_Wr, gat_br, gatout, HID / 2);

  // ---- ensemble ----
  k_ensemble<<<(N_NODES + TB - 1) / TB, TB, 0, stream>>>(gcnout, gatout, ens_W, ens_b, out);
}
